// MixtureLayer_15333033246888
// MI455X (gfx1250) — matrix operand order repaired and checked by reading, not compiled, not run on hardware
//
#include <hip/hip_runtime.h>
#include <math.h>

// ---- static problem dims (from reference) ----
#define DD     1024      // d_model
#define HH     4096      // ffn hidden
#define EE     8         // experts
#define GG     128       // groups
#define GSZ    128       // group size
#define CAPN   32        // capacity (slot 0 dropped -> 31 usable)
#define NTOK   16384     // B*S
#define HCHUNK 512       // H chunk held in LDS between the two GEMMs
#define SA     1032      // LDS A-tile row stride (bf16 elems), padded
#define SHS    520       // LDS H-tile row stride (bf16 elems), padded

typedef __bf16 bf16;
typedef __attribute__((ext_vector_type(16))) __bf16 v16bf;
typedef __attribute__((ext_vector_type(8)))  __bf16 bf16x8;
typedef __attribute__((ext_vector_type(4)))  __bf16 bf16x4;
typedef __attribute__((ext_vector_type(8)))  float  v8f;

union FragA { v16bf v; bf16x8 h[2]; };

__device__ __forceinline__ float gelu_tanh(float v) {
  float u = 0.7978845608028654f * (v + 0.044715f * v * v * v);
  return 0.5f * v * (1.0f + tanhf(u));
}

// ---------------------------------------------------------------------------
// fp32 -> bf16 straight convert (for x)
// ---------------------------------------------------------------------------
__global__ __launch_bounds__(256) void k_convert_x(const float* __restrict__ in,
                                                   bf16* __restrict__ out, int n4) {
  int i = blockIdx.x * blockDim.x + threadIdx.x;
  int stride = gridDim.x * blockDim.x;
  const float4* in4 = (const float4*)in;
  bf16x4* out4 = (bf16x4*)out;
  for (; i < n4; i += stride) {
    float4 v = in4[i];
    bf16x4 o;
    o[0] = (bf16)v.x; o[1] = (bf16)v.y; o[2] = (bf16)v.z; o[3] = (bf16)v.w;
    out4[i] = o;
  }
}

// ---------------------------------------------------------------------------
// fp32 -> bf16 tiled transpose: input element (r,c) at ip[r*inRow + c],
// output at op[c*outRow + r].  All dims multiples of 32 here.
// ---------------------------------------------------------------------------
__global__ __launch_bounds__(256) void k_transpose_bf16(
    const float* __restrict__ in, bf16* __restrict__ out,
    long inRow, long outRow, long inSlice, long outSlice) {
  __shared__ float tile[32][33];
  const float* ip = in + (long)blockIdx.z * inSlice;
  bf16* op = out + (long)blockIdx.z * outSlice;
  int c0 = blockIdx.x * 32, r0 = blockIdx.y * 32;
  int tx = threadIdx.x, ty = threadIdx.y;
#pragma unroll
  for (int j = 0; j < 32; j += 8)
    tile[ty + j][tx] = ip[(long)(r0 + ty + j) * inRow + (c0 + tx)];
  __syncthreads();
#pragma unroll
  for (int j = 0; j < 32; j += 8)
    op[(long)(c0 + ty + j) * outRow + (r0 + tx)] = (bf16)tile[tx][ty + j];
}

// ---------------------------------------------------------------------------
// Router: one block per group.  Computes softmax top-2 and replicates the
// reference per-k capacity cumsum, emitting slot->(token, gate) tables.
// ---------------------------------------------------------------------------
__global__ __launch_bounds__(128) void k_router(
    const float* __restrict__ x, const float* __restrict__ gate_w,
    const float* __restrict__ gate_b,
    int* __restrict__ slot_tok, float* __restrict__ slot_gate) {
  __shared__ float gw[DD * EE];
  __shared__ int   se[GSZ][2];
  __shared__ float sg[GSZ][2];
  int g = blockIdx.x, tid = threadIdx.x;
  for (int i = tid; i < DD * EE; i += 128) gw[i] = gate_w[i];
  // init this group's slot table
  long gbase = (long)g * EE * CAPN * 2;
  for (int i = tid; i < EE * CAPN * 2; i += 128) {
    slot_tok[gbase + i] = -1;
    slot_gate[gbase + i] = 0.f;
  }
  __syncthreads();

  float acc[EE];
#pragma unroll
  for (int e = 0; e < EE; ++e) acc[e] = gate_b[e];
  const float* xr = x + (long)(g * GSZ + tid) * DD;
  for (int d = 0; d < DD; ++d) {
    float xv = xr[d];
#pragma unroll
    for (int e = 0; e < EE; ++e) acc[e] += xv * gw[d * EE + e];
  }
  float m = acc[0];
#pragma unroll
  for (int e = 1; e < EE; ++e) m = fmaxf(m, acc[e]);
  float p[EE], s = 0.f;
#pragma unroll
  for (int e = 0; e < EE; ++e) { p[e] = expf(acc[e] - m); s += p[e]; }
  float inv = 1.f / s;
#pragma unroll
  for (int e = 0; e < EE; ++e) p[e] *= inv;
  int e0 = 0;
#pragma unroll
  for (int e = 1; e < EE; ++e) if (p[e] > p[e0]) e0 = e;
  int e1 = (e0 == 0) ? 1 : 0;
#pragma unroll
  for (int e = 0; e < EE; ++e) if (e != e0 && p[e] > p[e1]) e1 = e;
  se[tid][0] = e0; se[tid][1] = e1;
  sg[tid][0] = p[e0]; sg[tid][1] = p[e1];
  __syncthreads();

  if (tid == 0) {   // serial per-group capacity scan (256 trivial steps)
    int cnt[2][EE];
    for (int k = 0; k < 2; ++k)
      for (int e = 0; e < EE; ++e) cnt[k][e] = 0;
    for (int sI = 0; sI < GSZ; ++sI)
      for (int k = 0; k < 2; ++k) {
        int e = se[sI][k];
        int pos = ++cnt[k][e];
        if (pos < CAPN) {                       // valid, slot c = pos-1
          long idx = (((long)g * EE + e) * CAPN + (pos - 1)) * 2 + k;
          slot_tok[idx]  = sI;
          slot_gate[idx] = sg[sI][k];
        }
      }
  }
}

// ---------------------------------------------------------------------------
// Shared expert: fused x@W1 -> gelu -> @W2, 16-token tile per block.
// 8 waves: GEMM1 each wave owns 64 cols of the H chunk; GEMM2 each wave
// owns 128 cols of D.  Plain stores (initializes d_out).
// ---------------------------------------------------------------------------
__global__ __launch_bounds__(256) void k_shared_ffn(
    const bf16* __restrict__ xbf, const bf16* __restrict__ w1t,
    const bf16* __restrict__ w2t, const float* __restrict__ b1,
    const float* __restrict__ b2, float* __restrict__ out) {
  extern __shared__ char smem[];
  bf16* A  = (bf16*)smem;                       // 16 x SA
  bf16* Ht = (bf16*)(smem + 16 * SA * 2);       // 16 x SHS
  const int tid = threadIdx.x, wave = tid >> 5, lane = tid & 31;
  const int rsel = lane & 15, khi = lane >> 4;
  const long tokBase = (long)blockIdx.x * 16;
  const v8f vzero = {0.f,0.f,0.f,0.f,0.f,0.f,0.f,0.f};

  for (int i = tid; i < 16 * DD / 8; i += 256) {   // 16B copies
    int r = i >> 7, dd = (i & 127) * 8;
    *(uint4*)&A[r * SA + dd] = *(const uint4*)&xbf[(tokBase + r) * DD + dd];
  }
  __syncthreads();

  v8f Cacc[8];
#pragma unroll
  for (int nt = 0; nt < 8; ++nt) Cacc[nt] = vzero;

  for (int hc = 0; hc < HH; hc += HCHUNK) {
    v8f Hacc[4];
#pragma unroll
    for (int nt = 0; nt < 4; ++nt) Hacc[nt] = vzero;
    for (int k0 = 0; k0 < DD; k0 += 32) {
      FragA a;
      a.h[0] = *(const bf16x8*)&A[rsel * SA + k0 + khi * 16];
      a.h[1] = *(const bf16x8*)&A[rsel * SA + k0 + khi * 16 + 8];
#pragma unroll
      for (int nt = 0; nt < 4; ++nt) {
        int col = hc + wave * 64 + nt * 16 + rsel;
        FragA b; b.v = *(const v16bf*)(w1t + (long)col * DD + k0 + khi * 16);
        Hacc[nt] = __builtin_amdgcn_wmma_f32_16x16x32_bf16(
            false, a.v, false, b.v, (short)0, Hacc[nt], false, false);
      }
    }
#pragma unroll
    for (int nt = 0; nt < 4; ++nt) {
      int colL = wave * 64 + nt * 16 + rsel;
      float bb = b1[hc + colL];
#pragma unroll
      for (int i = 0; i < 8; ++i) {
        int row = i + ((lane < 16) ? 0 : 8);
        Ht[row * SHS + colL] = (bf16)gelu_tanh(Hacc[nt][i] + bb);
      }
    }
    __syncthreads();
    for (int k0 = 0; k0 < HCHUNK; k0 += 32) {
      FragA a;
      a.h[0] = *(const bf16x8*)&Ht[rsel * SHS + k0 + khi * 16];
      a.h[1] = *(const bf16x8*)&Ht[rsel * SHS + k0 + khi * 16 + 8];
#pragma unroll
      for (int nt = 0; nt < 8; ++nt) {
        int col = wave * 128 + nt * 16 + rsel;
        FragA b; b.v = *(const v16bf*)(w2t + (long)col * HH + hc + k0 + khi * 16);
        Cacc[nt] = __builtin_amdgcn_wmma_f32_16x16x32_bf16(
            false, a.v, false, b.v, (short)0, Cacc[nt], false, false);
      }
    }
    __syncthreads();
  }
#pragma unroll
  for (int nt = 0; nt < 8; ++nt) {
    int d = wave * 128 + nt * 16 + rsel;
    float bb = b2[d];
#pragma unroll
    for (int i = 0; i < 8; ++i) {
      int row = i + ((lane < 16) ? 0 : 8);
      out[(tokBase + row) * DD + d] = Cacc[nt][i] + bb;
    }
  }
}

// ---------------------------------------------------------------------------
// Expert FFN: block = (group g, expert e, capacity half).  Gather-sum slot
// rows into LDS, fused GEMM1->gelu->GEMM2, atomic scatter weighted by gates.
// ---------------------------------------------------------------------------
__global__ __launch_bounds__(256) void k_expert_ffn(
    const bf16* __restrict__ xbf, const bf16* __restrict__ w1t,
    const bf16* __restrict__ w2t, const float* __restrict__ key_b,
    const float* __restrict__ value_b, const int* __restrict__ slot_tok,
    const float* __restrict__ slot_gate, float* __restrict__ out) {
  extern __shared__ char smem[];
  bf16*  A   = (bf16*)smem;                              // 16 x SA
  bf16*  Ht  = (bf16*)(smem + 16 * SA * 2);              // 16 x SHS
  int*   st  = (int*)(smem + 16 * SA * 2 + 16 * SHS * 2);// 16 slots x 2
  float* sgt = (float*)(st + 32);
  const int g = blockIdx.x, e = blockIdx.y, half = blockIdx.z;
  const int tid = threadIdx.x, wave = tid >> 5, lane = tid & 31;
  const int rsel = lane & 15, khi = lane >> 4;
  const v8f vzero = {0.f,0.f,0.f,0.f,0.f,0.f,0.f,0.f};

  const long slotBase = (((long)g * EE + e) * CAPN + half * 16) * 2;
  if (tid < 32) { st[tid] = slot_tok[slotBase + tid]; sgt[tid] = slot_gate[slotBase + tid]; }
  __syncthreads();

  // gather-sum tokens into A tile (row c <- sum over its <=2 assigned tokens)
  for (int i = tid; i < 16 * DD; i += 256) {
    int c = i >> 10, d = i & (DD - 1);
    float v = 0.f;
    int t0 = st[c * 2 + 0];
    if (t0 >= 0) v += (float)xbf[((long)g * GSZ + t0) * DD + d];
    int t1 = st[c * 2 + 1];
    if (t1 >= 0) v += (float)xbf[((long)g * GSZ + t1) * DD + d];
    A[c * SA + d] = (bf16)v;
  }
  __syncthreads();

  const bf16* W1 = w1t + (long)e * HH * DD;   // [H][D] bf16 (transposed)
  const bf16* W2 = w2t + (long)e * DD * HH;   // [D][H] bf16 (transposed)

  v8f Cacc[8];
#pragma unroll
  for (int nt = 0; nt < 8; ++nt) Cacc[nt] = vzero;

  for (int hc = 0; hc < HH; hc += HCHUNK) {
    v8f Hacc[4];
#pragma unroll
    for (int nt = 0; nt < 4; ++nt) Hacc[nt] = vzero;
    for (int k0 = 0; k0 < DD; k0 += 32) {
      FragA a;
      a.h[0] = *(const bf16x8*)&A[rsel * SA + k0 + khi * 16];
      a.h[1] = *(const bf16x8*)&A[rsel * SA + k0 + khi * 16 + 8];
#pragma unroll
      for (int nt = 0; nt < 4; ++nt) {
        int col = hc + wave * 64 + nt * 16 + rsel;
        FragA b; b.v = *(const v16bf*)(W1 + (long)col * DD + k0 + khi * 16);
        Hacc[nt] = __builtin_amdgcn_wmma_f32_16x16x32_bf16(
            false, a.v, false, b.v, (short)0, Hacc[nt], false, false);
      }
    }
#pragma unroll
    for (int nt = 0; nt < 4; ++nt) {
      int colL = wave * 64 + nt * 16 + rsel;
      float bb = key_b[(long)e * HH + hc + colL];
#pragma unroll
      for (int i = 0; i < 8; ++i) {
        int row = i + ((lane < 16) ? 0 : 8);
        Ht[row * SHS + colL] = (bf16)gelu_tanh(Hacc[nt][i] + bb);
      }
    }
    __syncthreads();
    for (int k0 = 0; k0 < HCHUNK; k0 += 32) {
      FragA a;
      a.h[0] = *(const bf16x8*)&Ht[rsel * SHS + k0 + khi * 16];
      a.h[1] = *(const bf16x8*)&Ht[rsel * SHS + k0 + khi * 16 + 8];
#pragma unroll
      for (int nt = 0; nt < 8; ++nt) {
        int col = wave * 128 + nt * 16 + rsel;
        FragA b; b.v = *(const v16bf*)(W2 + (long)col * HH + hc + k0 + khi * 16);
        Cacc[nt] = __builtin_amdgcn_wmma_f32_16x16x32_bf16(
            false, a.v, false, b.v, (short)0, Cacc[nt], false, false);
      }
    }
    __syncthreads();
  }

  // scatter: out[token] += gate * (eo + value_bias)
#pragma unroll
  for (int nt = 0; nt < 8; ++nt) {
    int d = wave * 128 + nt * 16 + rsel;
    float vb = value_b[(long)e * DD + d];
#pragma unroll
    for (int i = 0; i < 8; ++i) {
      int c = i + ((lane < 16) ? 0 : 8);
      float val = Cacc[nt][i] + vb;
      int t0 = st[c * 2 + 0];
      if (t0 >= 0) atomicAdd(&out[((long)g * GSZ + t0) * DD + d], sgt[c * 2 + 0] * val);
      int t1 = st[c * 2 + 1];
      if (t1 >= 0) atomicAdd(&out[((long)g * GSZ + t1) * DD + d], sgt[c * 2 + 1] * val);
    }
  }
}

// ---------------------------------------------------------------------------
extern "C" void kernel_launch(void* const* d_in, const int* in_sizes, int n_in,
                              void* d_out, int out_size, void* d_ws, size_t ws_size,
                              hipStream_t stream) {
  (void)in_sizes; (void)n_in; (void)out_size; (void)ws_size;
  const float* x         = (const float*)d_in[0];
  const float* gate_w    = (const float*)d_in[1];
  const float* gate_b    = (const float*)d_in[2];
  const float* keys      = (const float*)d_in[3];
  const float* key_b     = (const float*)d_in[4];
  const float* values    = (const float*)d_in[5];
  const float* value_b   = (const float*)d_in[6];
  const float* s_keys    = (const float*)d_in[7];
  const float* s_key_b   = (const float*)d_in[8];
  const float* s_values  = (const float*)d_in[9];
  const float* s_value_b = (const float*)d_in[10];
  float* out = (float*)d_out;

  char* ws = (char*)d_ws;
  size_t off = 0;
  auto alloc = [&](size_t bytes) -> void* {
    void* p = ws + off;
    off += (bytes + 255) & ~(size_t)255;
    return p;
  };
  bf16*  xbf       = (bf16*)alloc((size_t)NTOK * DD * 2);
  bf16*  w1t       = (bf16*)alloc((size_t)EE * HH * DD * 2);   // [e][h][d]
  bf16*  w2t       = (bf16*)alloc((size_t)EE * DD * HH * 2);   // [e][d][h]
  bf16*  sw1t      = (bf16*)alloc((size_t)HH * DD * 2);        // [h][d]
  bf16*  sw2t      = (bf16*)alloc((size_t)DD * HH * 2);        // [d][h]
  int*   slot_tok  = (int*)alloc((size_t)GG * EE * CAPN * 2 * sizeof(int));
  float* slot_gate = (float*)alloc((size_t)GG * EE * CAPN * 2 * sizeof(float));

  // 1) precision conversion + weight transposes (one pass; bf16 expert
  //    weights = 128 MB -> resident in 192 MB L2 for the GEMM phase)
  k_convert_x<<<2048, 256, 0, stream>>>(x, xbf, NTOK * DD / 4);
  dim3 tb(32, 8);
  // keys[d][e][h] -> w1t[e][h][d]
  k_transpose_bf16<<<dim3(HH / 32, DD / 32, EE), tb, 0, stream>>>(
      keys, w1t, (long)EE * HH, (long)DD, (long)HH, (long)HH * DD);
  // values[h][e][d] -> w2t[e][d][h]
  k_transpose_bf16<<<dim3(DD / 32, HH / 32, EE), tb, 0, stream>>>(
      values, w2t, (long)EE * DD, (long)HH, (long)DD, (long)DD * HH);
  // s_keys[d][0][h] -> sw1t[h][d]
  k_transpose_bf16<<<dim3(HH / 32, DD / 32, 1), tb, 0, stream>>>(
      s_keys, sw1t, (long)HH, (long)DD, 0, 0);
  // s_values[h][0][d] -> sw2t[d][h]
  k_transpose_bf16<<<dim3(DD / 32, HH / 32, 1), tb, 0, stream>>>(
      s_values, sw2t, (long)DD, (long)HH, 0, 0);

  // 2) routing (independent of conversions)
  k_router<<<GG, 128, 0, stream>>>(x, gate_w, gate_b, slot_tok, slot_gate);

  // 3) shared expert writes out (plain stores), then expert FFN atomically adds
  size_t smem_sh = (size_t)16 * SA * 2 + (size_t)16 * SHS * 2;
  size_t smem_ex = smem_sh + 32 * sizeof(int) + 32 * sizeof(float);
  k_shared_ffn<<<NTOK / 16, 256, smem_sh, stream>>>(xbf, sw1t, sw2t,
                                                    s_key_b, s_value_b, out);
  k_expert_ffn<<<dim3(GG, EE, 2), 256, smem_ex, stream>>>(
      xbf, w1t, w2t, key_b, value_b, slot_tok, slot_gate, out);
}
